// DIAMNet_53206054863225
// MI455X (gfx1250) — matrix operand order repaired and checked by reading, not compiled, not run on hardware
//
#include <hip/hip_runtime.h>

typedef __bf16 bf16;
typedef __attribute__((ext_vector_type(16))) __bf16 v16bf;
typedef __attribute__((ext_vector_type(8)))  float  v8f;

#define NHEADS 4
#define DMODEL 256
#define BATCH  16
#define PLEN_  512
#define GLEN_  16384
#define MEMLEN 16

union Frag { v16bf v; uint4 q[2]; };

// ---------------------------------------------------------------------------
// mem0 = segment mean of graph over 1024-row segments
// ---------------------------------------------------------------------------
__global__ __launch_bounds__(256)
void pool_kernel(const float* __restrict__ graph, float* __restrict__ mem) {
    const int bm = blockIdx.x;          // b*16 + m
    const int d  = threadIdx.x;
    const float* src = graph + (size_t)bm * 1024u * 256u + d;
    float s = 0.f;
    for (int k = 0; k < 1024; k++) s += src[(size_t)k * 256u];
    mem[(size_t)bm * 256u + d] = s * (1.0f / 1024.0f);
}

// ---------------------------------------------------------------------------
// Y[M,256] = X[M,256] @ W[256,256], bf16 WMMA, f32 accumulate.
// Block: 256 thr = 8 waves; M_BLK=64 (wm=wave&3 -> 16 rows), N split 2x128
// (wn=wave>>2, 8 16x16 tiles per wave). K loop 8 x 32.
// ---------------------------------------------------------------------------
template <bool OUT_F32>
__global__ __launch_bounds__(256)
void gemm256_kernel(const float* __restrict__ X, const float* __restrict__ W,
                    bf16* __restrict__ Ybf, float* __restrict__ Yf) {
    __shared__ __align__(16) bf16 As[64][40];    // padded rows: 80B stride
    __shared__ __align__(16) bf16 Bs[256][40];   // stored [n][k]
    const int t    = threadIdx.x;
    const int wave = t >> 5, lane = t & 31;
    const int wm = wave & 3, wn = wave >> 2;
    const int hlf = lane >> 4, lr = lane & 15;
    const int mbase = blockIdx.x * 64;

    v8f acc[8] = {};
    for (int kk = 0; kk < 256; kk += 32) {
        // A: 64x32 f32 -> bf16 (coalesced)
#pragma unroll
        for (int i = 0; i < 8; i++) {
            int c = i * 256 + t, r = c >> 5, k = c & 31;
            As[r][k] = (bf16)X[(size_t)(mbase + r) * 256u + kk + k];
        }
        // B: W[kk+k][n] -> Bs[n][k] (coalesced read, LDS transpose)
#pragma unroll
        for (int i = 0; i < 32; i++) {
            int c = i * 256 + t, k = c >> 8, n = c & 255;
            Bs[n][k] = (bf16)W[(size_t)(kk + k) * 256u + n];
        }
        __syncthreads();
        Frag a;
        a.q[0] = *reinterpret_cast<const uint4*>(&As[wm * 16 + lr][hlf * 16]);
        a.q[1] = *reinterpret_cast<const uint4*>(&As[wm * 16 + lr][hlf * 16 + 8]);
#pragma unroll
        for (int nt = 0; nt < 8; nt++) {
            Frag bf_;
            const bf16* bp = &Bs[wn * 128 + nt * 16 + lr][hlf * 16];
            bf_.q[0] = reinterpret_cast<const uint4*>(bp)[0];
            bf_.q[1] = reinterpret_cast<const uint4*>(bp)[1];
            acc[nt] = __builtin_amdgcn_wmma_f32_16x16x32_bf16(
                false, a.v, false, bf_.v, (short)0, acc[nt], false, false);
        }
        __syncthreads();
    }
#pragma unroll
    for (int nt = 0; nt < 8; nt++) {
        int n = wn * 128 + nt * 16 + lr;
#pragma unroll
        for (int v = 0; v < 8; v++) {
            int row = mbase + wm * 16 + v + hlf * 8;   // C: M = v (+8 for hi half)
            if (OUT_F32) Yf [(size_t)row * 256u + n] = acc[nt][v];
            else         Ybf[(size_t)row * 256u + n] = (bf16)acc[nt][v];
        }
    }
}

// ---------------------------------------------------------------------------
// scores[b,h,i,j] = scale * hq(16,64) . hk(j,64)  (masked)
// grid (B*H, klen/256); each of 8 waves does 2 j-tiles; K=64 in two 32-chunks
// ---------------------------------------------------------------------------
__global__ __launch_bounds__(256)
void score_kernel(const bf16* __restrict__ hq, const bf16* __restrict__ hk,
                  const int* __restrict__ mask, float* __restrict__ S, int klen) {
    const int t = threadIdx.x, wave = t >> 5, lane = t & 31;
    const int hlf = lane >> 4, lr = lane & 15;
    const int b = blockIdx.x >> 2, h = blockIdx.x & 3;

    Frag a0, a1;
    {   // A = hq rows (row = lane%16), elements: 16 contiguous K per half-wave
        const bf16* qrow = hq + (size_t)(b * 16 + lr) * 256u + h * 64;
        a0.q[0] = *reinterpret_cast<const uint4*>(qrow + hlf * 16);
        a0.q[1] = *reinterpret_cast<const uint4*>(qrow + hlf * 16 + 8);
        a1.q[0] = *reinterpret_cast<const uint4*>(qrow + 32 + hlf * 16);
        a1.q[1] = *reinterpret_cast<const uint4*>(qrow + 40 + hlf * 16);
    }
    float* srow = S + (size_t)(b * 4 + h) * 16u * (size_t)klen;
#pragma unroll
    for (int jt2 = 0; jt2 < 2; jt2++) {
        const int j16 = blockIdx.y * 16 + wave * 2 + jt2;
        const int jb  = j16 * 16;
        const int j   = jb + lr;                 // B: col = lane%16
        const bf16* krow = hk + ((size_t)b * klen + j) * 256u + h * 64;
        Frag b0, b1;                             // elements: 16 contiguous K per half
        b0.q[0] = reinterpret_cast<const uint4*>(krow + hlf * 16)[0];
        b0.q[1] = reinterpret_cast<const uint4*>(krow + hlf * 16)[1];
        b1.q[0] = reinterpret_cast<const uint4*>(krow + 32 + hlf * 16)[0];
        b1.q[1] = reinterpret_cast<const uint4*>(krow + 32 + hlf * 16)[1];
        v8f acc = {};
        acc = __builtin_amdgcn_wmma_f32_16x16x32_bf16(false, a0.v, false, b0.v, (short)0, acc, false, false);
        acc = __builtin_amdgcn_wmma_f32_16x16x32_bf16(false, a1.v, false, b1.v, (short)0, acc, false, false);
        const int km = mask[b * klen + j];
#pragma unroll
        for (int v = 0; v < 8; v++) {
            int i = v + hlf * 8;
            srow[(size_t)i * klen + j] = km ? acc[v] * 0.125f : -1e30f;
        }
    }
}

// ---------------------------------------------------------------------------
// softmax over j per row; writes bf16 probabilities
// ---------------------------------------------------------------------------
__global__ __launch_bounds__(256)
void softmax_kernel(const float* __restrict__ S, bf16* __restrict__ P, int klen) {
    __shared__ float red[256];
    const int t = threadIdx.x;
    const float* s = S + (size_t)blockIdx.x * klen;
    bf16* p = P + (size_t)blockIdx.x * klen;
    float m = -3.4e38f;
    for (int j = t; j < klen; j += 256) m = fmaxf(m, s[j]);
    red[t] = m; __syncthreads();
    for (int o = 128; o > 0; o >>= 1) { if (t < o) red[t] = fmaxf(red[t], red[t + o]); __syncthreads(); }
    m = red[0]; __syncthreads();
    float sum = 0.f;
    for (int j = t; j < klen; j += 256) sum += __expf(s[j] - m);
    red[t] = sum; __syncthreads();
    for (int o = 128; o > 0; o >>= 1) { if (t < o) red[t] += red[t + o]; __syncthreads(); }
    const float inv = 1.0f / fmaxf(red[0], 1e-30f);
    for (int j = t; j < klen; j += 256) p[j] = (bf16)(__expf(s[j] - m) * inv);
}

// ---------------------------------------------------------------------------
// vec[b,i,h*64+d] = sum_j P[b,h,i,j] * hv[b,j,h*64+d]
// grid = B*H; 8 waves split j (K dim); hv tile transposed through LDS so
// B-fragments are contiguous b128 reads; partial tiles reduced via LDS atomics
// ---------------------------------------------------------------------------
__global__ __launch_bounds__(256)
void attnv_kernel(const bf16* __restrict__ P, const bf16* __restrict__ hv,
                  float* __restrict__ vec, int klen) {
    __shared__ __align__(16) bf16 hvT[8][64][40];   // [wave][d][j], padded
    __shared__ float vtile[16][64];
    const int t = threadIdx.x, wave = t >> 5, lane = t & 31;
    const int hlf = lane >> 4, lr = lane & 15;
    const int b = blockIdx.x >> 2, h = blockIdx.x & 3;

    for (int c = t; c < 1024; c += 256) vtile[c >> 6][c & 63] = 0.f;
    __syncthreads();

    v8f acc[4] = {};
    const int span = klen >> 3;
    const int j0 = wave * span;
    const bf16* prow = P + (size_t)(b * 4 + h) * 16u * (size_t)klen;

    for (int jb = j0; jb < j0 + span; jb += 32) {
        {   // stage hv[jb+lane][h*64 .. +63] and transpose into LDS
            const bf16* src = hv + ((size_t)b * klen + jb + lane) * 256u + h * 64;
            uint4 rowdat[4];
            rowdat[0] = reinterpret_cast<const uint4*>(src)[0];
            rowdat[1] = reinterpret_cast<const uint4*>(src)[1];
            rowdat[2] = reinterpret_cast<const uint4*>(src)[2];
            rowdat[3] = reinterpret_cast<const uint4*>(src)[3];
            const bf16* rb = reinterpret_cast<const bf16*>(rowdat);
#pragma unroll
            for (int d = 0; d < 64; d++) hvT[wave][d][lane] = rb[d];
        }
        __syncthreads();
        Frag a;   // A = P tile, row = lane%16, K = j contiguous per half
        a.q[0] = *reinterpret_cast<const uint4*>(prow + (size_t)lr * klen + jb + hlf * 16);
        a.q[1] = *reinterpret_cast<const uint4*>(prow + (size_t)lr * klen + jb + hlf * 16 + 8);
#pragma unroll
        for (int dt = 0; dt < 4; dt++) {
            Frag bf_; // B: col = d (lane%16), K = j contiguous per half
            const bf16* bp = &hvT[wave][dt * 16 + lr][hlf * 16];
            bf_.q[0] = reinterpret_cast<const uint4*>(bp)[0];
            bf_.q[1] = reinterpret_cast<const uint4*>(bp)[1];
            acc[dt] = __builtin_amdgcn_wmma_f32_16x16x32_bf16(
                false, a.v, false, bf_.v, (short)0, acc[dt], false, false);
        }
        __syncthreads();
    }
#pragma unroll
    for (int dt = 0; dt < 4; dt++)
#pragma unroll
        for (int v = 0; v < 8; v++)
            atomicAdd(&vtile[v + hlf * 8][dt * 16 + lr], acc[dt][v]);
    __syncthreads();
    for (int c = t; c < 1024; c += 256) {
        int i = c >> 6, d = c & 63;
        vec[(size_t)(b * 16 + i) * 256u + h * 64 + d] = vtile[i][d];
    }
}

// ---------------------------------------------------------------------------
// out = g*q + (1-g)*attn_out, g = sigmoid([q,attn_out] @ Wg + bg)
// ---------------------------------------------------------------------------
__global__ __launch_bounds__(256)
void gate_kernel(const float* __restrict__ q, const float* __restrict__ ao,
                 const float* __restrict__ Wg, const float* __restrict__ bg,
                 float* __restrict__ out) {
    const int row = blockIdx.x, n = threadIdx.x;
    const float* qr = q  + (size_t)row * 256u;
    const float* ar = ao + (size_t)row * 256u;
    float acc = bg[n];
    for (int k = 0; k < 256; k++) acc += qr[k] * Wg[(size_t)k * 256u + n];
    for (int k = 0; k < 256; k++) acc += ar[k] * Wg[(size_t)(256 + k) * 256u + n];
    const float g = 1.0f / (1.0f + __expf(-acc));
    out[(size_t)row * 256u + n] = g * qr[n] + (1.0f - g) * ar[n];
}

// ---------------------------------------------------------------------------
extern "C" void kernel_launch(void* const* d_in, const int* in_sizes, int n_in,
                              void* d_out, int out_size, void* d_ws, size_t ws_size,
                              hipStream_t stream) {
    (void)in_sizes; (void)n_in; (void)out_size; (void)ws_size;
    const float* pattern = (const float*)d_in[0];
    const float* graph   = (const float*)d_in[1];
    const int*   pmask   = (const int*)d_in[2];
    const int*   gmask   = (const int*)d_in[3];
    const float* pW[6];  for (int i = 0; i < 6; i++) pW[i] = (const float*)d_in[4 + i];
    const float* gW[6];  for (int i = 0; i < 6; i++) gW[i] = (const float*)d_in[10 + i];

    // workspace carve (256B-aligned)
    char* w = (char*)d_ws;
    auto alloc = [&](size_t bytes) -> char* {
        char* p = w; w += (bytes + 255) & ~(size_t)255; return p;
    };
    float* mem_a = (float*)alloc(65536u * 4);
    float* mem_b = (float*)alloc(65536u * 4);
    bf16*  hq    = (bf16*) alloc(65536u * 2);
    float* vec   = (float*)alloc(65536u * 4);
    float* aout  = (float*)alloc(65536u * 4);
    bf16*  p_hk  = (bf16*) alloc((size_t)BATCH * PLEN_ * 256u * 2);
    bf16*  p_hv  = (bf16*) alloc((size_t)BATCH * PLEN_ * 256u * 2);
    bf16*  g_hk  = (bf16*) alloc((size_t)BATCH * GLEN_ * 256u * 2);
    bf16*  g_hv  = (bf16*) alloc((size_t)BATCH * GLEN_ * 256u * 2);
    float* S     = (float*)alloc((size_t)BATCH * NHEADS * 16u * GLEN_ * 4);
    bf16*  P     = (bf16*) alloc((size_t)BATCH * NHEADS * 16u * GLEN_ * 2);

    // init mem + hoisted (loop-invariant) K/V projections
    pool_kernel<<<BATCH * MEMLEN, 256, 0, stream>>>(graph, mem_a);
    gemm256_kernel<false><<<BATCH * PLEN_ / 64, 256, 0, stream>>>(pattern, pW[1], p_hk, nullptr);
    gemm256_kernel<false><<<BATCH * PLEN_ / 64, 256, 0, stream>>>(pattern, pW[2], p_hv, nullptr);
    gemm256_kernel<false><<<BATCH * GLEN_ / 64, 256, 0, stream>>>(graph,   gW[1], g_hk, nullptr);
    gemm256_kernel<false><<<BATCH * GLEN_ / 64, 256, 0, stream>>>(graph,   gW[2], g_hv, nullptr);

    float* cur = mem_a;
    float* nxt = mem_b;
    for (int step = 0; step < 3; step++) {
        // ---- attention vs pattern ----
        gemm256_kernel<false><<<4, 256, 0, stream>>>(cur, pW[0], hq, nullptr);
        score_kernel<<<dim3(BATCH * NHEADS, PLEN_ / 256), 256, 0, stream>>>(hq, p_hk, pmask, S, PLEN_);
        softmax_kernel<<<BATCH * NHEADS * 16, 256, 0, stream>>>(S, P, PLEN_);
        attnv_kernel<<<BATCH * NHEADS, 256, 0, stream>>>(P, p_hv, vec, PLEN_);
        gemm256_kernel<true><<<4, 256, 0, stream>>>(vec, pW[3], nullptr, aout);
        gate_kernel<<<BATCH * 16, 256, 0, stream>>>(cur, aout, pW[4], pW[5], nxt);
        { float* tmp = cur; cur = nxt; nxt = tmp; }
        // ---- attention vs graph ----
        gemm256_kernel<false><<<4, 256, 0, stream>>>(cur, gW[0], hq, nullptr);
        score_kernel<<<dim3(BATCH * NHEADS, GLEN_ / 256), 256, 0, stream>>>(hq, g_hk, gmask, S, GLEN_);
        softmax_kernel<<<BATCH * NHEADS * 16, 256, 0, stream>>>(S, P, GLEN_);
        attnv_kernel<<<BATCH * NHEADS, 256, 0, stream>>>(P, g_hv, vec, GLEN_);
        gemm256_kernel<true><<<4, 256, 0, stream>>>(vec, gW[3], nullptr, aout);
        float* dst = (step == 2) ? (float*)d_out : nxt;
        gate_kernel<<<BATCH * 16, 256, 0, stream>>>(cur, aout, gW[4], gW[5], dst);
        { float* tmp = cur; cur = nxt; nxt = tmp; }
        cur = (step == 2) ? (float*)d_out : cur;  // (unused after final step)
    }
}
